// GraphLearningProp_53807350284661
// MI455X (gfx1250) — compile-verified, matrix operand order checked by reading, not run on hardware
//
#include <hip/hip_runtime.h>
#include <hip/hip_bf16.h>

typedef __attribute__((ext_vector_type(2)))  float  v2f;
typedef __attribute__((ext_vector_type(8)))  float  v8f;
typedef __attribute__((ext_vector_type(8)))  __bf16 v8bf;
typedef __attribute__((ext_vector_type(16))) __bf16 v16bf;

#define BDIM 4096
#define CDIM 256
#define KNN  10

static constexpr float EPS_  = 0.5f;
static constexpr float LAM_  = 0.2f;
static constexpr float BETA_ = 1.0f;

// ---- order-preserving float <-> uint mapping for atomic min/max ----
__device__ __forceinline__ unsigned f2ord(float f) {
    unsigned u = __float_as_uint(f);
    return (u & 0x80000000u) ? ~u : (u | 0x80000000u);
}
__device__ __forceinline__ float ord2f(unsigned u) {
    return (u & 0x80000000u) ? __uint_as_float(u & 0x7fffffffu) : __uint_as_float(~u);
}

// ---- per-row min/max reduce of a wave's 16x16 WMMA tile, then atomics ----
__device__ __forceinline__ void tile_row_minmax(const v8f& acc, int rowBase, int lane,
                                                unsigned* __restrict__ minK,
                                                unsigned* __restrict__ maxK) {
    const int h = lane >> 4;
#pragma unroll
    for (int v = 0; v < 8; ++v) {
        int row = rowBase + v + 8 * h;
        float mn = acc[v], mx = acc[v];
#pragma unroll
        for (int m = 1; m < 16; m <<= 1) {
            mn = fminf(mn, __shfl_xor(mn, m, 32));
            mx = fmaxf(mx, __shfl_xor(mx, m, 32));
        }
        if ((lane & 15) == 0) {
            atomicMin(&minK[row], f2ord(mn));
            atomicMax(&maxK[row], f2ord(mx));
        }
    }
}

// ================= FP32 WMMA gram: G = F * F^T, store + row min/max ========
__global__ __launch_bounds__(256) void gram_f32_wmma(
    const float* __restrict__ F, float* __restrict__ G,
    unsigned* __restrict__ minK, unsigned* __restrict__ maxK)
{
    __shared__ float As[32][66];
    __shared__ float Bs[64][66];
    const int tid  = threadIdx.x;
    const int lane = tid & 31;
    const int wave = tid >> 5;
    const int wr   = wave >> 2;   // 0..1  (16-row tile within 32)
    const int wc   = wave & 3;    // 0..3  (16-col tile within 64)
    const int i0   = blockIdx.y * 32;
    const int j0   = blockIdx.x * 64;
    const int h    = lane >> 4;
    const int l16  = lane & 15;

    v8f acc = {};
    for (int kc = 0; kc < CDIM; kc += 64) {
        for (int idx = tid; idx < 32 * 32; idx += 256) {      // A: 32 rows x 64 cols (float2)
            int r = idx >> 5, c2 = idx & 31;
            v2f val = *(const v2f*)(F + (size_t)(i0 + r) * CDIM + kc + 2 * c2);
            *(v2f*)&As[r][2 * c2] = val;
        }
        for (int idx = tid; idx < 64 * 32; idx += 256) {      // B: 64 rows x 64 cols
            int r = idx >> 5, c2 = idx & 31;
            v2f val = *(const v2f*)(F + (size_t)(j0 + r) * CDIM + kc + 2 * c2);
            *(v2f*)&Bs[r][2 * c2] = val;
        }
        __syncthreads();
#pragma unroll
        for (int kk = 0; kk < 64; kk += 4) {
            // A 16x4 f32 layout: lane -> M = lane%16, VGPRs hold K = 2*(lane/16)+{0,1}
            v2f a = *(const v2f*)&As[16 * wr + l16][kk + 2 * h];
            v2f b = *(const v2f*)&Bs[16 * wc + l16][kk + 2 * h];
            acc = __builtin_amdgcn_wmma_f32_16x16x4_f32(false, a, false, b,
                                                        (short)0, acc, false, false);
        }
        __syncthreads();
    }
    const int col = j0 + 16 * wc + l16;
#pragma unroll
    for (int v = 0; v < 8; ++v) {
        int row = i0 + 16 * wr + v + 8 * h;       // C/D layout: VGPR v -> M = v + 8*(lane/16)
        G[(size_t)row * BDIM + col] = acc[v];
    }
    tile_row_minmax(acc, i0 + 16 * wr, lane, minK, maxK);
}

// ============ BF16 WMMA gram (streaming): row min/max only, no store =======
// Tile staging uses CDNA5 async global->LDS DMA (GLOBAL_LOAD_ASYNC_TO_LDS_B128,
// ASYNCcnt-tracked, no VGPR round trip), GVS mode: SGPR base + 32-bit lane offset.
__global__ __launch_bounds__(256) void gram_bf16_wmma(
    const __bf16* __restrict__ F,
    unsigned* __restrict__ minK, unsigned* __restrict__ maxK)
{
    __shared__ __bf16 As[32][72];   // row stride 144B -> every 16B slot aligned
    __shared__ __bf16 Bs[64][72];
    const int tid  = threadIdx.x;
    const int lane = tid & 31;
    const int wave = tid >> 5;
    const int wr   = wave >> 2;
    const int wc   = wave & 3;
    const int i0   = blockIdx.y * 32;
    const int j0   = blockIdx.x * 64;
    const int h    = lane >> 4;
    const int l16  = lane & 15;

    // Per-thread 16B staging slots: thread -> (row, 8-elem column group)
    const int rr  = tid >> 3;                 // 0..31
    const int cg8 = (tid & 7) * 8;            // col 0,8,...,56
    // LDS byte offsets (addr[31:0] of the generic pointer == LDS offset)
    const unsigned ldsA  = (unsigned)(uintptr_t)&As[rr][cg8];
    const unsigned ldsB0 = (unsigned)(uintptr_t)&Bs[rr][cg8];
    const unsigned ldsB1 = (unsigned)(uintptr_t)&Bs[32 + rr][cg8];

    v8f acc = {};
    for (int kc = 0; kc < CDIM; kc += 64) {
        const unsigned offA  = (unsigned)(((i0 + rr)      * CDIM + kc + cg8) * 2);
        const unsigned offB0 = (unsigned)(((j0 + rr)      * CDIM + kc + cg8) * 2);
        const unsigned offB1 = (unsigned)(((j0 + 32 + rr) * CDIM + kc + cg8) * 2);
        asm volatile("global_load_async_to_lds_b128 %0, %1, %2"
                     :: "v"(ldsA),  "v"(offA),  "s"(F) : "memory");
        asm volatile("global_load_async_to_lds_b128 %0, %1, %2"
                     :: "v"(ldsB0), "v"(offB0), "s"(F) : "memory");
        asm volatile("global_load_async_to_lds_b128 %0, %1, %2"
                     :: "v"(ldsB1), "v"(offB1), "s"(F) : "memory");
        asm volatile("s_wait_asynccnt 0x0" ::: "memory");
        __syncthreads();
#pragma unroll
        for (int kk = 0; kk < 64; kk += 32) {
            // 16-bit A 16x32 layout: lane -> M=lane%16; K chunks [8h,8h+8) and [16+8h,+8)
            v8bf alo = *(const v8bf*)&As[16 * wr + l16][kk + 8 * h];
            v8bf ahi = *(const v8bf*)&As[16 * wr + l16][kk + 16 + 8 * h];
            v8bf blo = *(const v8bf*)&Bs[16 * wc + l16][kk + 8 * h];
            v8bf bhi = *(const v8bf*)&Bs[16 * wc + l16][kk + 16 + 8 * h];
            v16bf a, b;
#pragma unroll
            for (int e = 0; e < 8; ++e) { a[e] = alo[e]; a[e + 8] = ahi[e];
                                          b[e] = blo[e]; b[e + 8] = bhi[e]; }
            acc = __builtin_amdgcn_wmma_f32_16x16x32_bf16(false, a, false, b,
                                                          (short)0, acc, false, false);
        }
        __syncthreads();
    }
    tile_row_minmax(acc, i0 + 16 * wr, lane, minK, maxK);
}

// =========================== small helper kernels ==========================
__global__ void init_setup(unsigned* gmn, unsigned* gmx, float* gAcc) {
    int i = blockIdx.x * 256 + threadIdx.x;
    gmn[i] = 0xFFFFFFFFu; gmx[i] = 0u;
    if (i == 0) gAcc[0] = 0.f;
}
__global__ void init_minmax(unsigned* mn, unsigned* mx) {
    int i = blockIdx.x * 256 + threadIdx.x;
    mn[i] = 0xFFFFFFFFu; mx[i] = 0u;
}

__global__ __launch_bounds__(256) void rownorm(const float* __restrict__ H, float* __restrict__ xx) {
    __shared__ float red[256];
    int i = blockIdx.x, t = threadIdx.x;
    float v = H[(size_t)i * CDIM + t];
    red[t] = v * v; __syncthreads();
    for (int s = 128; s > 0; s >>= 1) { if (t < s) red[t] += red[t + s]; __syncthreads(); }
    if (t == 0) xx[i] = red[0];
}

__device__ __forceinline__ float d1_value(float g, float xi, float xj, float lij,
                                          float gmn, float ginv) {
    float d2 = xi + xj - 2.f * g;
    return sqrtf(fmaxf(d2, 1e-12f)) - 2.f * BETA_ * lij - 1e-5f * ((g - gmn) * ginv);
}

__global__ __launch_bounds__(256) void d1_minmax(
    const float* __restrict__ G, const float* __restrict__ L, const float* __restrict__ xx,
    const unsigned* __restrict__ gminK, const unsigned* __restrict__ gmaxK,
    float* __restrict__ d1min, float* __restrict__ d1max)
{
    __shared__ float smn[256], smx[256];
    int i = blockIdx.x, t = threadIdx.x;
    float gmn = ord2f(gminK[i]);
    float ginv = 1.f / (ord2f(gmaxK[i]) - gmn + 1e-8f);
    float xi = xx[i];
    float mn = 3.4e38f, mx = -3.4e38f;
    for (int j = t; j < BDIM; j += 256) {
        float d1 = d1_value(G[(size_t)i * BDIM + j], xi, xx[j], L[(size_t)i * BDIM + j], gmn, ginv);
        mn = fminf(mn, d1); mx = fmaxf(mx, d1);
    }
    smn[t] = mn; smx[t] = mx; __syncthreads();
    for (int s = 128; s > 0; s >>= 1) {
        if (t < s) { smn[t] = fminf(smn[t], smn[t + s]); smx[t] = fmaxf(smx[t], smx[t + s]); }
        __syncthreads();
    }
    if (t == 0) { d1min[i] = smn[0]; d1max[i] = smx[0]; }
}

// ===== stable top-12 per row (values + indices), emit nbr/D_nbr/dk terms ===
__global__ __launch_bounds__(256) void topk_kernel(
    const float* __restrict__ G, const float* __restrict__ L, const float* __restrict__ xx,
    const unsigned* __restrict__ gminK, const unsigned* __restrict__ gmaxK,
    const float* __restrict__ d1min, const float* __restrict__ d1max,
    int* __restrict__ nbr, float* __restrict__ Dnbr,
    float* __restrict__ dksum, float* __restrict__ gAcc)
{
    __shared__ float cv[256 * 12];
    __shared__ int   cj[256 * 12];
    __shared__ float rv[256]; __shared__ int rj[256]; __shared__ int ro[256];
    __shared__ float resv[12]; __shared__ int resj[12]; __shared__ int reso;
    int i = blockIdx.x, t = threadIdx.x;
    float gmn = ord2f(gminK[i]);
    float ginv = 1.f / (ord2f(gmaxK[i]) - gmn + 1e-8f);
    float xi = xx[i];
    float dmn = d1min[i];
    float dinv = 1.f / (d1max[i] - dmn + 1e-8f);

    float av[12]; int aj[12];
#pragma unroll
    for (int p = 0; p < 12; ++p) { av[p] = 3.4e38f; aj[p] = 0x7fffffff; }
    for (int j = t; j < BDIM; j += 256) {
        float d1 = d1_value(G[(size_t)i * BDIM + j], xi, xx[j], L[(size_t)i * BDIM + j], gmn, ginv);
        float D = fmaxf((d1 - dmn) * dinv, 0.f);
        if (D < av[11] || (D == av[11] && j < aj[11])) {
            int p = 11;
            while (p > 0 && (D < av[p - 1] || (D == av[p - 1] && j < aj[p - 1]))) {
                av[p] = av[p - 1]; aj[p] = aj[p - 1]; --p;
            }
            av[p] = D; aj[p] = j;
        }
    }
#pragma unroll
    for (int p = 0; p < 12; ++p) { cv[t * 12 + p] = av[p]; cj[t * 12 + p] = aj[p]; }
    __syncthreads();

    int h = 0;                                    // per-thread head into its sorted list
    for (int r = 0; r < 12; ++r) {
        rv[t] = (h < 12) ? cv[t * 12 + h] : 3.4e38f;
        rj[t] = (h < 12) ? cj[t * 12 + h] : 0x7fffffff;
        ro[t] = t;
        __syncthreads();
        for (int s = 128; s > 0; s >>= 1) {
            if (t < s) {
                float ov = rv[t + s]; int oj = rj[t + s];
                if (ov < rv[t] || (ov == rv[t] && oj < rj[t])) {
                    rv[t] = ov; rj[t] = oj; ro[t] = ro[t + s];
                }
            }
            __syncthreads();
        }
        if (t == 0) { resv[r] = rv[0]; resj[r] = rj[0]; reso = ro[0]; }
        __syncthreads();
        if (t == reso) ++h;
        __syncthreads();
    }
    if (t == 0) {
        float s = 0.f;
        for (int k = 1; k <= KNN; ++k) {
            nbr[i * KNN + k - 1] = resj[k];
            Dnbr[i * KNN + k - 1] = resv[k];
            s += resv[k];
        }
        dksum[i] = s;
        atomicAdd(gAcc, 0.5f * ((float)KNN * resv[11] - s));
    }
}

__global__ void eta_kernel(const float* __restrict__ dksum, const float* __restrict__ gAcc,
                           float* __restrict__ eta) {
    int i = blockIdx.x * 256 + threadIdx.x;
    float gamma = gAcc[0] / (float)BDIM;
    eta[i] = (1.f / (float)KNN) * (1.f + dksum[i] / (2.f * gamma + 1e-8f));
}

// ============ one propagation step: row-wise dots, gate, mix ===============
__global__ __launch_bounds__(256) void update_kernel(
    const float* __restrict__ FmIn, const float* __restrict__ H,
    const unsigned* __restrict__ minK, const unsigned* __restrict__ maxK,
    const int* __restrict__ nbr, const float* __restrict__ Dnbr,
    const float* __restrict__ eta, const float* __restrict__ gAcc,
    float* __restrict__ FmOut, __bf16* __restrict__ FmOutBf)
{
    __shared__ float red[256];
    __shared__ float Ak[KNN];
    __shared__ int   nb[KNN];
    int i = blockIdx.x, t = threadIdx.x;
    if (t < KNN) nb[t] = nbr[i * KNN + t];
    __syncthreads();
    float fic = FmIn[(size_t)i * CDIM + t];
    float hic = H[(size_t)i * CDIM + t];
    float amn = ord2f(minK[i]);
    float arng = 1.f / (ord2f(maxK[i]) - amn + 1e-8f);
    float gamma = gAcc[0] / (float)BDIM;
    float inv2g = 1.f / (2.f * gamma + 1e-8f);
    float eti = eta[i];
    for (int k = 0; k < KNN; ++k) {
        red[t] = fic * FmIn[(size_t)nb[k] * CDIM + t];
        __syncthreads();
        for (int s = 128; s > 0; s >>= 1) { if (t < s) red[t] += red[t + s]; __syncthreads(); }
        if (t == 0) {
            float attn = (red[0] - amn) * arng;
            float dis  = Dnbr[i * KNN + k] - LAM_ * attn;
            Ak[k] = fmaxf(eti - dis * inv2g, 0.f);
        }
        __syncthreads();
    }
    float out = 0.f;
#pragma unroll
    for (int k = 0; k < KNN; ++k) out += Ak[k] * H[(size_t)nb[k] * CDIM + t];
    float res = EPS_ * out + (1.f - EPS_) * hic;
    FmOut[(size_t)i * CDIM + t] = res;
    FmOutBf[(size_t)i * CDIM + t] = (__bf16)res;
}

// ============================== launch =====================================
extern "C" void kernel_launch(void* const* d_in, const int* in_sizes, int n_in,
                              void* d_out, int out_size, void* d_ws, size_t ws_size,
                              hipStream_t stream)
{
    (void)in_sizes; (void)n_in; (void)out_size; (void)ws_size;
    const float* H = (const float*)d_in[0];   // (1,B,C)
    const float* L = (const float*)d_in[1];   // (B,B)
    float* out = (float*)d_out;

    char* ws = (char*)d_ws;
    size_t off = 0;
    auto alloc = [&](size_t bytes) -> void* {
        void* p = ws + off;
        off += (bytes + 255) & ~(size_t)255;
        return p;
    };
    float*    G     = (float*)alloc((size_t)BDIM * BDIM * 4);
    float*    fm0   = (float*)alloc((size_t)BDIM * CDIM * 4);
    float*    fm1   = (float*)alloc((size_t)BDIM * CDIM * 4);
    __bf16*   bf0   = (__bf16*)alloc((size_t)BDIM * CDIM * 2);
    __bf16*   bf1   = (__bf16*)alloc((size_t)BDIM * CDIM * 2);
    float*    xx    = (float*)alloc(BDIM * 4);
    unsigned* gmnK  = (unsigned*)alloc(BDIM * 4);
    unsigned* gmxK  = (unsigned*)alloc(BDIM * 4);
    unsigned* amnK  = (unsigned*)alloc(BDIM * 4);
    unsigned* amxK  = (unsigned*)alloc(BDIM * 4);
    float*    d1mn  = (float*)alloc(BDIM * 4);
    float*    d1mx  = (float*)alloc(BDIM * 4);
    int*      nbrA  = (int*)alloc((size_t)BDIM * KNN * 4);
    float*    DnbrA = (float*)alloc((size_t)BDIM * KNN * 4);
    float*    dksum = (float*)alloc(BDIM * 4);
    float*    etaA  = (float*)alloc(BDIM * 4);
    float*    gAcc  = (float*)alloc(256);

    dim3 blk(256);
    dim3 gramGrid(BDIM / 64, BDIM / 32);

    init_setup<<<BDIM / 256, blk, 0, stream>>>(gmnK, gmxK, gAcc);
    rownorm<<<BDIM, blk, 0, stream>>>(H, xx);
    gram_f32_wmma<<<gramGrid, blk, 0, stream>>>(H, G, gmnK, gmxK);
    d1_minmax<<<BDIM, blk, 0, stream>>>(G, L, xx, gmnK, gmxK, d1mn, d1mx);
    topk_kernel<<<BDIM, blk, 0, stream>>>(G, L, xx, gmnK, gmxK, d1mn, d1mx,
                                          nbrA, DnbrA, dksum, gAcc);
    eta_kernel<<<BDIM / 256, blk, 0, stream>>>(dksum, gAcc, etaA);

    // t = 0: Fm == H, attn min/max == setup gram min/max -> skip one full gram
    update_kernel<<<BDIM, blk, 0, stream>>>(H, H, gmnK, gmxK, nbrA, DnbrA, etaA, gAcc, fm0, bf0);
    // t = 1
    init_minmax<<<BDIM / 256, blk, 0, stream>>>(amnK, amxK);
    gram_bf16_wmma<<<gramGrid, blk, 0, stream>>>(bf0, amnK, amxK);
    update_kernel<<<BDIM, blk, 0, stream>>>(fm0, H, amnK, amxK, nbrA, DnbrA, etaA, gAcc, fm1, bf1);
    // t = 2 (final -> d_out)
    init_minmax<<<BDIM / 256, blk, 0, stream>>>(amnK, amxK);
    gram_bf16_wmma<<<gramGrid, blk, 0, stream>>>(bf1, amnK, amxK);
    update_kernel<<<BDIM, blk, 0, stream>>>(fm1, H, amnK, amxK, nbrA, DnbrA, etaA, gAcc, out, bf0);
}
